// SwarmSetEquivariantActorCritic_28028956574056
// MI455X (gfx1250) — matrix operand order repaired and checked by reading, not compiled, not run on hardware
//
#include <hip/hip_runtime.h>

// ---------------------------------------------------------------------------
// SwarmSetEquivariantActorCritic on gfx1250 (MI455X), wave32 + WMMA bf16.
// One block (128 threads = 4 waves) processes a 16-agent tile. All dense
// matmuls -> v_wmma_f32_16x16x32_bf16 (f32 accum). Activations that feed
// WMMA A-operands live in LDS as bf16 (converted once at store); fragment
// loads are 2x b128 per lane. Weights pre-transposed to bf16 [N][Kpad] in
// d_ws by a prep kernel.
// ---------------------------------------------------------------------------

#define NBN     65536      // B*N agents
#define D_OBS   182
#define NWAVES  4

typedef __attribute__((ext_vector_type(16))) __bf16 v16bf;
typedef __attribute__((ext_vector_type(8)))  float  v8f;

// ---------------- scalar helpers ----------------
__device__ __forceinline__ unsigned short f32_to_bf16(float f) {
  unsigned u = __float_as_uint(f);
  unsigned r = (u + 0x7FFFu + ((u >> 16) & 1u)) >> 16;   // round-nearest-even
  return (unsigned short)r;
}
__device__ __forceinline__ float bf2f(unsigned short s) {
  return __uint_as_float(((unsigned)s) << 16);
}
__device__ __forceinline__ float bf16rt(float x) { return bf2f(f32_to_bf16(x)); }
__device__ __forceinline__ float gelu_f(float x) {
  // 0.5*x*(1+tanh(u)) == x * (1 - 1/(exp(2u)+1)), u = sqrt(2/pi)*(x+0.044715x^3)
  float u = 0.7978845608028654f * (x + 0.044715f * x * x * x);
  float e = __expf(2.f * u);
  return x * (1.f - 1.f / (e + 1.f));
}
__device__ __forceinline__ float sigm(float x) { return 1.f / (1.f + __expf(-x)); }
__device__ __forceinline__ float tanh_f(float x) {
  float e = __expf(2.f * x);
  return 1.f - 2.f / (e + 1.f);
}

// ---------------- WMMA fragment loaders ----------------
// A-matrix 16x32 bf16 from LDS. Per ISA: lane m=L&15, hi=L>>4; per lane the
// 16 K-values are two contiguous 8-element (16B) spans at K = hi*8 and 16+hi*8.
__device__ __forceinline__ v16bf load_a(const unsigned short* A, int lda, int lane) {
  int m = lane & 15, hi = lane >> 4;
  const unsigned short* row = A + m * lda + hi * 8;
  union { v16bf v; uint4 q[2]; } r;
  r.q[0] = *(const uint4*)(row);
  r.q[1] = *(const uint4*)(row + 16);
  return r.v;
}
// B-matrix 32x16 bf16 from global, weight transposed [N][Kpad]. Per lane the
// 16 K-values are two 8-element spans at K = k0+hi*8 and k0+16+hi*8 (matches A).
__device__ __forceinline__ v16bf load_b(const unsigned short* WT, int kpad,
                                        int n0, int k0, int lane) {
  int n  = n0 + (lane & 15);
  int kb = k0 + ((lane >> 4) << 3);
  const uint4* p = (const uint4*)(WT + (size_t)n * kpad + kb);
  union { v16bf v; uint4 q[2]; } r;
  r.q[0] = p[0];
  r.q[1] = p[2];
  return r.v;
}
__device__ __forceinline__ v8f wmma_bf16(v16bf a, v16bf b, v8f c) {
  return __builtin_amdgcn_wmma_f32_16x16x32_bf16(false, a, false, b, (short)0, c, false, false);
}

// ---------------- tiled GEMM: C = act(A @ W^T + bias) ----------------------
// A: bf16 LDS. Output modes:
//   0: f32, none            1: bf16, gelu           2: f32, gelu*rowscale[m]
//   3: f32, (x)*scl         4: f32, C+=x            5: f32, C=(C+x)*rowscale[m]
//   6: bf16, none
__device__ void gemm_x(const unsigned short* __restrict__ A, int lda,
                       const unsigned short* __restrict__ WT, int kpad,
                       const float* __restrict__ bias,
                       void* __restrict__ Cv, int ldc,
                       int Mtiles, int Ntiles, int Ktiles,
                       int mode, const float* rowscale, float scl,
                       int wave, int lane) {
  float* Cf = (float*)Cv;
  unsigned short* Cb = (unsigned short*)Cv;
  int jobs = Mtiles * Ntiles;
  for (int t = wave; t < jobs; t += NWAVES) {
    int mt = t / Ntiles, nt = t - mt * Ntiles;
    v8f acc = {0.f, 0.f, 0.f, 0.f, 0.f, 0.f, 0.f, 0.f};
    const unsigned short* Abase = A + (mt * 16) * lda;
    for (int kt = 0; kt < Ktiles; ++kt) {
      v16bf av = load_a(Abase + kt * 32, lda, lane);
      v16bf bv = load_b(WT, kpad, nt * 16, kt * 32, lane);
      acc = wmma_bf16(av, bv, acc);
    }
    int n  = nt * 16 + (lane & 15);
    int mb = mt * 16 + ((lane >> 4) << 3);
    float bn = bias ? bias[n] : 0.f;
#pragma unroll
    for (int vv = 0; vv < 8; ++vv) {
      int m = mb + vv;
      float x = acc[vv] + bn;
      if (mode == 1)      Cb[m * ldc + n] = f32_to_bf16(gelu_f(x));
      else if (mode == 6) Cb[m * ldc + n] = f32_to_bf16(x);
      else {
        float* dst = Cf + m * ldc + n;
        if (mode == 0)      *dst = x;
        else if (mode == 2) *dst = gelu_f(x) * rowscale[m];
        else if (mode == 3) *dst = x * scl;
        else if (mode == 4) *dst = *dst + x;
        else                *dst = (*dst + x) * rowscale[m];
      }
    }
  }
}

// ---------------- workspace layout (bf16 elems, transposed [N][Kpad]) ------
#define OW_TOK  0        /* 64 x 32   */
#define OW_SQ   2048     /* 64 x 64   */
#define OW_SK   6144
#define OW_SV   10240
#define OW_SO   14336
#define OW_M1   18432    /* 128 x 64  */
#define OW_M2   26624    /* 128 x 128 */
#define OW_M3   43008    /* 64 x 128  */
#define OW_E1   51200    /* 128 x 32  */
#define OW_E2   55296
#define OW_E3   71680    /* 64 x 128  */
#define OW_L1   79872
#define OW_L2   96256
#define OW_PK   112640   /* 64 x 64   */
#define OW_PV   116736
#define OW_PO   120832
#define OW_TER  124928   /* 16 x 160  */
#define OW_PAIR 127488   /* 64 x 32   */
#define OW_H1   129536   /* 128 x 288 */
#define OW_H2   166400
#define OW_H3   182784
#define OW_END  199168
#define POOLQ_BYTE_OFF (OW_END * 2)     // f32[64] after bf16 region

// ---------------- prep kernel: f32 -> bf16 transposed + pool query --------
struct PrepArgs {
  const float *tok_w, *s_wq, *s_wk, *s_wv, *s_wo;
  const float *m1w1, *m1w2, *m1w3;
  const float *ew1, *ew2, *ew3;
  const float *lw1, *lw2;
  const float *p_wk, *p_wv, *p_wo, *p_wq, *p_bq, *seed;
  const float *terw, *pairw, *hw1, *hw2, *hw3;
  unsigned short* W;
  float* poolq;
};
__device__ void convT(const float* __restrict__ src, unsigned short* __restrict__ dst,
                      int K, int N, int Kpad, int gt, int gn) {
  int tot = N * Kpad;
  for (int i = gt; i < tot; i += gn) {
    int n = i / Kpad, k = i - n * Kpad;
    float f = (k < K) ? src[(size_t)k * N + n] : 0.f;
    dst[i] = f32_to_bf16(f);
  }
}
__global__ void swarm_prep_kernel(PrepArgs A) {
  int gt = blockIdx.x * blockDim.x + threadIdx.x;
  int gn = gridDim.x * blockDim.x;
  convT(A.tok_w, A.W + OW_TOK, 6, 64, 32, gt, gn);
  convT(A.s_wq,  A.W + OW_SQ, 64, 64, 64, gt, gn);
  convT(A.s_wk,  A.W + OW_SK, 64, 64, 64, gt, gn);
  convT(A.s_wv,  A.W + OW_SV, 64, 64, 64, gt, gn);
  convT(A.s_wo,  A.W + OW_SO, 64, 64, 64, gt, gn);
  convT(A.m1w1,  A.W + OW_M1, 64, 128, 64, gt, gn);
  convT(A.m1w2,  A.W + OW_M2, 128, 128, 128, gt, gn);
  convT(A.m1w3,  A.W + OW_M3, 128, 64, 128, gt, gn);
  convT(A.ew1,   A.W + OW_E1, 16, 128, 32, gt, gn);
  convT(A.ew2,   A.W + OW_E2, 128, 128, 128, gt, gn);
  convT(A.ew3,   A.W + OW_E3, 128, 64, 128, gt, gn);
  convT(A.lw1,   A.W + OW_L1, 128, 128, 128, gt, gn);
  convT(A.lw2,   A.W + OW_L2, 128, 128, 128, gt, gn);
  convT(A.p_wk,  A.W + OW_PK, 64, 64, 64, gt, gn);
  convT(A.p_wv,  A.W + OW_PV, 64, 64, 64, gt, gn);
  convT(A.p_wo,  A.W + OW_PO, 64, 64, 64, gt, gn);
  convT(A.terw,  A.W + OW_TER, 144, 16, 160, gt, gn);
  convT(A.pairw, A.W + OW_PAIR, 10, 64, 32, gt, gn);
  convT(A.hw1,   A.W + OW_H1, 288, 128, 288, gt, gn);
  convT(A.hw2,   A.W + OW_H2, 128, 128, 128, gt, gn);
  convT(A.hw3,   A.W + OW_H3, 128, 128, 128, gt, gn);
  if (gt < 64) {                 // pool query: (seed @ wq + bq) / sqrt(DH)
    float s = 0.f;
    for (int k = 0; k < 64; ++k) s += A.seed[k] * A.p_wq[k * 64 + gt];
    A.poolq[gt] = (s + A.p_bq[gt]) * 0.25f;
  }
}

// ---------------- LDS arena (float units; bf16 regions noted) --------------
#define O_TOK   0        /* f32 [80][64] residual x                           */
#define O_LN    5120     /* bf16 [80][64]: tokA / LN out / attn out / tok-copy/ hA */
#define O_Q     7680     /* f32 [80][64] q, poolK; later t1 bf16 region       */
#define O_K     12800    /* f32 [80][64] k, poolV                             */
#define O_V     17920    /* f32 [80][64] v                                    */
#define O_H1    23040    /* bf16 [16][128]                                    */
#define O_H2    24064    /* bf16 [16][128]                                    */
#define O_LA    25088    /* bf16 [16][128] staging / pool attn out            */
#define O_MF    26112    /* f32 [80] mask                                     */
#define O_EGO   26192    /* bf16 [16][32]                                     */
#define O_ECTX  26448    /* bf16 [16][64] ego_ctx, later pool c               */
#define O_LOG   26960    /* f32 [80]                                          */
#define O_ALPHA 27040    /* f32 [80]                                          */
#define O_VR    27120    /* f32 [16][3]                                       */
#define O_VU    27168    /* f32 [16][3]                                       */
#define O_TERT  27216    /* bf16 [16][144]                                    */
#define O_PE    28368    /* bf16 [16][64]                                     */
#define O_HID   28880    /* f32 [16][128]                                     */
#define S_TOTAL 30928    /* 123712 bytes                                      */

struct KArgs {
  const float* obs;
  const unsigned short* W;
  const float* poolq;
  const float *tok_b, *ln1_s, *ln1_b, *ln2_s, *ln2_b;
  const float *sbq, *sbk, *sbv, *sbo;
  const float *m1b1, *m1b2, *m1b3;
  const float *eb1, *eb2, *eb3;
  const float *lb1, *lb2, *lb3, *lw3;
  const float *pbk, *pbv, *pbo;
  const float *conv_k, *conv_b, *ter_b;
  const float *pair_b;
  const float *hb1, *hb2, *hb3;
  const float *gate_w, *gate_b, *mag_w, *mag_b, *lstd_w, *lstd_b, *val_w, *val_b;
  float* out;
};

// LayerNorm over 64 cols for 80 rows; f32 in, bf16 out.
__device__ void ln80(const float* X, unsigned short* Y,
                     const float* s, const float* b, int tid) {
  if (tid < 80) {
    const float* x = X + tid * 64;
    unsigned short* y = Y + tid * 64;
    float m = 0.f;
    for (int c = 0; c < 64; ++c) m += x[c];
    m *= (1.f / 64.f);
    float v = 0.f;
    for (int c = 0; c < 64; ++c) { float d = x[c] - m; v += d * d; }
    v *= (1.f / 64.f);
    float r = rsqrtf(v + 1e-6f);
    for (int c = 0; c < 64; ++c) y[c] = f32_to_bf16((x[c] - m) * r * s[c] + b[c]);
  }
}

__global__ void __launch_bounds__(128, 1)
swarm_actor_critic_kernel(KArgs P) {
  __shared__ float S[S_TOTAL];
  const int tid  = threadIdx.x;
  const int wave = tid >> 5;
  const int lane = tid & 31;
  const int abase = blockIdx.x * 16;
  const float* obsB = P.obs + (size_t)abase * D_OBS;
  const unsigned short* W = P.W;

  unsigned short* sLN = (unsigned short*)(S + O_LN);   // 5120 bf16
  unsigned short* sH1 = (unsigned short*)(S + O_H1);
  unsigned short* sH2 = (unsigned short*)(S + O_H2);
  unsigned short* sLA = (unsigned short*)(S + O_LA);
  unsigned short* sEGO  = (unsigned short*)(S + O_EGO);
  unsigned short* sECTX = (unsigned short*)(S + O_ECTX);
  unsigned short* sTERT = (unsigned short*)(S + O_TERT);
  unsigned short* sPE   = (unsigned short*)(S + O_PE);

  for (int i = tid; i < 16; i += 128) __builtin_prefetch(obsB + i * D_OBS, 0, 0);

  // ---- S0: token inputs [inv(4),fre,tem] -> tokA bf16 [80][32] @ sLN; mask -
  for (int i = tid; i < 80; i += 128) {
    int s = i / 16, a = i - s * 16;
    const float* nb = obsB + a * D_OBS + 16 + s * 15;
    S[O_MF + i] = (fabsf(nb[0]) > 1e-6f || fabsf(nb[1]) > 1e-6f || fabsf(nb[2]) > 1e-6f)
                      ? 1.f : 0.f;
  }
  for (int i = tid; i < 80 * 32; i += 128) {
    int row = i >> 5, col = i & 31;
    int s = row / 16, a = row - s * 16;
    const float* nb = obsB + a * D_OBS + 16 + s * 15;
    float v = 0.f;
    if (col < 4)       v = nb[11 + col];
    else if (col == 4) v = nb[9];
    else if (col == 5) v = nb[10];
    sLN[i] = f32_to_bf16(v);
  }
  __syncthreads();

  // ---- S1: tok(f32) = gelu(tokA @ tok_w + b) * mf ----
  gemm_x(sLN, 32, W + OW_TOK, 32, P.tok_b, S + O_TOK, 64, 5, 4, 1, 2, S + O_MF, 0.f, wave, lane);
  __syncthreads();

  // ---- S2: LN1 -> sLN (bf16) ----
  ln80(S + O_TOK, sLN, P.ln1_s, P.ln1_b, tid);
  __syncthreads();

  // ---- S3: q (scaled 1/sqrt(16)), k, v -> f32 ----
  gemm_x(sLN, 64, W + OW_SQ, 64, P.sbq, S + O_Q, 64, 5, 4, 2, 3, nullptr, 0.25f, wave, lane);
  gemm_x(sLN, 64, W + OW_SK, 64, P.sbk, S + O_K, 64, 5, 4, 2, 0, nullptr, 0.f, wave, lane);
  gemm_x(sLN, 64, W + OW_SV, 64, P.sbv, S + O_V, 64, 5, 4, 2, 0, nullptr, 0.f, wave, lane);
  __syncthreads();

  // ---- S4: self-attention, one (agent,head) per thread; out -> sLN bf16 ----
  if (tid < 64) {
    int a = tid & 15, h = tid >> 4;
    for (int qi = 0; qi < 5; ++qi) {
      float sc[5], mx = -3.0e38f;
      for (int kj = 0; kj < 5; ++kj) {
        float d = 0.f;
        const float* qp = S + O_Q + (qi * 16 + a) * 64 + h * 16;
        const float* kp = S + O_K + (kj * 16 + a) * 64 + h * 16;
        for (int dd = 0; dd < 16; ++dd) d += qp[dd] * kp[dd];
        if (S[O_MF + kj * 16 + a] == 0.f) d = -1e9f;
        sc[kj] = d;
        mx = fmaxf(mx, d);
      }
      float se = 0.f;
      for (int kj = 0; kj < 5; ++kj) { sc[kj] = __expf(sc[kj] - mx); se += sc[kj]; }
      float inv = 1.f / se;
      for (int dd = 0; dd < 16; ++dd) {
        float o = 0.f;
        for (int kj = 0; kj < 5; ++kj) o += sc[kj] * S[O_V + (kj * 16 + a) * 64 + h * 16 + dd];
        sLN[(qi * 16 + a) * 64 + h * 16 + dd] = f32_to_bf16(o * inv);
      }
    }
  }
  __syncthreads();

  // ---- S5: out-proj + residual: tok += attn @ wo + bo ----
  gemm_x(sLN, 64, W + OW_SO, 64, P.sbo, S + O_TOK, 64, 5, 4, 2, 4, nullptr, 0.f, wave, lane);
  __syncthreads();

  // ---- S6: LN2 then per-token MLP, tok = (tok + mlp) * mf ----
  ln80(S + O_TOK, sLN, P.ln2_s, P.ln2_b, tid);
  __syncthreads();
  for (int s = 0; s < 5; ++s) {
    gemm_x(sLN + s * 1024, 64, W + OW_M1, 64, P.m1b1, sH1, 128, 1, 8, 2, 1, nullptr, 0.f, wave, lane);
    __syncthreads();
    gemm_x(sH1, 128, W + OW_M2, 128, P.m1b2, sH2, 128, 1, 8, 4, 1, nullptr, 0.f, wave, lane);
    __syncthreads();
    gemm_x(sH2, 128, W + OW_M3, 128, P.m1b3, S + O_TOK + s * 1024, 64, 1, 4, 4, 5, S + O_MF + s * 16, 0.f, wave, lane);
    __syncthreads();
  }

  // ---- tok-copy: bf16 image of final tok for later A-operands -> sLN ----
  for (int i = tid; i < 80 * 64; i += 128) sLN[i] = f32_to_bf16(S[O_TOK + i]);
  __syncthreads();

  // ---- S7: ego MLP -> ego_ctx bf16 [16][64] ----
  for (int i = tid; i < 16 * 32; i += 128) {
    int row = i >> 5, col = i & 31;
    sEGO[i] = (col < 16) ? f32_to_bf16(obsB[row * D_OBS + col]) : (unsigned short)0;
  }
  __syncthreads();
  gemm_x(sEGO, 32, W + OW_E1, 32, P.eb1, sH1, 128, 1, 8, 1, 1, nullptr, 0.f, wave, lane);
  __syncthreads();
  gemm_x(sH1, 128, W + OW_E2, 128, P.eb2, sH2, 128, 1, 8, 4, 1, nullptr, 0.f, wave, lane);
  __syncthreads();
  gemm_x(sH2, 128, W + OW_E3, 128, P.eb3, sECTX, 64, 1, 4, 4, 6, nullptr, 0.f, wave, lane);
  __syncthreads();

  // ---- S8: per-token logits MLP, masked softmax alpha, v_r / v_u ----
  for (int s = 0; s < 5; ++s) {
    for (int i = tid; i < 16 * 128; i += 128) {
      int row = i >> 7, col = i & 127;
      sLA[i] = (col < 64) ? sLN[(s * 16 + row) * 64 + col]
                          : sECTX[row * 64 + (col - 64)];
    }
    __syncthreads();
    gemm_x(sLA, 128, W + OW_L1, 128, P.lb1, sH1, 128, 1, 8, 4, 1, nullptr, 0.f, wave, lane);
    __syncthreads();
    gemm_x(sH1, 128, W + OW_L2, 128, P.lb2, sH2, 128, 1, 8, 4, 1, nullptr, 0.f, wave, lane);
    __syncthreads();
    if (tid < 16) {
      const unsigned short* h2 = sH2 + tid * 128;
      float acc = P.lb3[0];
      for (int c = 0; c < 128; ++c) acc += bf2f(h2[c]) * P.lw3[c];
      S[O_LOG + s * 16 + tid] = acc;
    }
    __syncthreads();
  }
  if (tid < 16) {
    int a = tid;
    float ml[5], mx = -3.0e38f;
    for (int s = 0; s < 5; ++s) {
      float m = S[O_MF + s * 16 + a];
      ml[s] = (m != 0.f) ? S[O_LOG + s * 16 + a] : -1e9f;
      mx = fmaxf(mx, ml[s]);
    }
    float ex[5], se = 0.f;
    for (int s = 0; s < 5; ++s) { ex[s] = __expf(ml[s] - mx) * S[O_MF + s * 16 + a]; se += ex[s]; }
    float inv = 1.f / (se + 1e-9f);
    for (int s = 0; s < 5; ++s) S[O_ALPHA + s * 16 + a] = ex[s] * inv;
    for (int d = 0; d < 3; ++d) {
      float ar = 0.f, au = 0.f;
      for (int s = 0; s < 5; ++s) {
        const float* nb = obsB + a * D_OBS + 16 + s * 15;
        float al = S[O_ALPHA + s * 16 + a];
        ar += al * nb[d];
        au += al * nb[3 + d];
      }
      S[O_VR + a * 3 + d] = ar;
      S[O_VU + a * 3 + d] = au;
    }
  }
  __syncthreads();

  // ---- S9: pooling attention (precomputed query) -> c bf16 [16][64] ----
  gemm_x(sLN, 64, W + OW_PK, 64, P.pbk, S + O_Q, 64, 5, 4, 2, 0, nullptr, 0.f, wave, lane);
  gemm_x(sLN, 64, W + OW_PV, 64, P.pbv, S + O_K, 64, 5, 4, 2, 0, nullptr, 0.f, wave, lane);
  __syncthreads();
  if (tid < 64) {
    int a = tid & 15, h = tid >> 4;
    float sc[5], mx = -3.0e38f;
    for (int kj = 0; kj < 5; ++kj) {
      float d = 0.f;
      const float* kp = S + O_Q + (kj * 16 + a) * 64 + h * 16;
      for (int dd = 0; dd < 16; ++dd) d += P.poolq[h * 16 + dd] * kp[dd];
      if (S[O_MF + kj * 16 + a] == 0.f) d = -1e9f;
      sc[kj] = d;
      mx = fmaxf(mx, d);
    }
    float se = 0.f;
    for (int kj = 0; kj < 5; ++kj) { sc[kj] = __expf(sc[kj] - mx); se += sc[kj]; }
    float inv = 1.f / se;
    for (int dd = 0; dd < 16; ++dd) {
      float o = 0.f;
      for (int kj = 0; kj < 5; ++kj) o += sc[kj] * S[O_K + (kj * 16 + a) * 64 + h * 16 + dd];
      sLA[a * 64 + h * 16 + dd] = f32_to_bf16(o * inv);
    }
  }
  __syncthreads();
  gemm_x(sLA, 64, W + OW_PO, 64, P.pbo, sECTX, 64, 1, 4, 2, 6, nullptr, 0.f, wave, lane);
  __syncthreads();

  // ---- S10: terrain conv (bf16) + second GEMM -> tert bf16 [16][144] ----
  {
    unsigned short* t1 = (unsigned short*)(S + O_Q);   // [(y*16+a)][160] bf16
    for (int i = tid; i < 16 * 9 * 160; i += 128) t1[i] = 0;
    __syncthreads();
    for (int i = tid; i < 16 * 81; i += 128) {
      int a = i / 81, cell = i - a * 81, y = cell / 9, x = cell - y * 9;
      float val[9];
#pragma unroll
      for (int k = 0; k < 9; ++k) {
        int dy = k / 3, dx = k - dy * 3;
        int rr = y + dy, cc = x + dx;
        float nv = 0.f;
        if (cc >= 1 && cc <= 9) {
          int ry = (rr == 0) ? 8 : ((rr == 10) ? 0 : rr - 1);  // row wrap, col zero-pad
          nv = obsB[a * D_OBS + 101 + ry * 9 + (cc - 1)];
        }
        val[k] = bf16rt(nv);
      }
      for (int c = 0; c < 16; ++c) {
        float acc = 0.f;
        for (int k = 0; k < 9; ++k) acc += val[k] * bf16rt(P.conv_k[k * 16 + c]);
        float o = gelu_f(acc + bf16rt(P.conv_b[c]));
        t1[(y * 16 + a) * 160 + x * 16 + c] = f32_to_bf16(o);
      }
    }
    __syncthreads();
    // [144 rows (y*16+a)] x [K=144 pad 160] @ ter_w^T [16][160]
    const unsigned short* t1c = (const unsigned short*)(S + O_Q);
    for (int t = wave; t < 9; t += NWAVES) {
      v8f acc = {0.f, 0.f, 0.f, 0.f, 0.f, 0.f, 0.f, 0.f};
      for (int kt = 0; kt < 5; ++kt) {
        v16bf av = load_a(t1c + (t * 16) * 160 + kt * 32, 160, lane);
        v16bf bv = load_b(W + OW_TER, 160, 0, kt * 32, lane);
        acc = wmma_bf16(av, bv, acc);
      }
      int n  = lane & 15;
      int mb = (lane >> 4) << 3;
      float bn = P.ter_b[n];
#pragma unroll
      for (int vv = 0; vv < 8; ++vv) {
        int a = mb + vv;                       // local row == agent
        sTERT[a * 144 + t * 16 + n] = f32_to_bf16(gelu_f(acc[vv] + bn));
      }
    }
    __syncthreads();
  }

  // ---- S11: pair embedding -> bf16 [16][64] ----
  for (int i = tid; i < 16 * 32; i += 128) {
    int row = i >> 5, col = i & 31;
    sLA[i] = (col < 10) ? f32_to_bf16(obsB[row * D_OBS + 91 + col]) : (unsigned short)0;
  }
  __syncthreads();
  gemm_x(sLA, 32, W + OW_PAIR, 32, P.pair_b, sPE, 64, 1, 4, 1, 1, nullptr, 0.f, wave, lane);
  __syncthreads();

  // ---- S12: h_mlp on concat [ego|c|pair_emb|tert] (288) ----
  for (int i = tid; i < 16 * 288; i += 128) {
    int row = i / 288, col = i - row * 288;
    unsigned short v;
    if (col < 16)       v = f32_to_bf16(obsB[row * D_OBS + col]);
    else if (col < 80)  v = sECTX[row * 64 + (col - 16)];
    else if (col < 144) v = sPE[row * 64 + (col - 80)];
    else                v = sTERT[row * 144 + (col - 144)];
    sLN[i] = v;
  }
  __syncthreads();
  gemm_x(sLN, 288, W + OW_H1, 288, P.hb1, sH1, 128, 1, 8, 9, 1, nullptr, 0.f, wave, lane);
  __syncthreads();
  gemm_x(sH1, 128, W + OW_H2, 128, P.hb2, sH2, 128, 1, 8, 4, 1, nullptr, 0.f, wave, lane);
  __syncthreads();
  gemm_x(sH2, 128, W + OW_H3, 128, P.hb3, S + O_HID, 128, 1, 8, 4, 0, nullptr, 0.f, wave, lane);
  __syncthreads();

  // ---- S13: heads ----
  if (tid < 16) {
    int a = tid;
    const float* h = S + O_HID + a * 128;
    float g0 = P.gate_b[0], g1 = P.gate_b[1], mg = P.mag_b[0], vl = P.val_b[0];
    float l0 = P.lstd_b[0], l1 = P.lstd_b[1], l2 = P.lstd_b[2];
    for (int c = 0; c < 128; ++c) {
      float hv = h[c];
      g0 += hv * P.gate_w[c * 2 + 0];
      g1 += hv * P.gate_w[c * 2 + 1];
      mg += hv * P.mag_w[c];
      vl += hv * P.val_w[c];
      l0 += hv * P.lstd_w[c * 3 + 0];
      l1 += hv * P.lstd_w[c * 3 + 1];
      l2 += hv * P.lstd_w[c * 3 + 2];
    }
    g0 = sigm(g0); g1 = sigm(g1); mg = sigm(mg);
    float vx[3];
    for (int d = 0; d < 3; ++d)
      vx[d] = g0 * S[O_VR + a * 3 + d] + g1 * S[O_VU + a * 3 + d];
    float nrm = sqrtf(vx[0] * vx[0] + vx[1] * vx[1] + vx[2] * vx[2]) + 1e-6f;
    size_t g = (size_t)abase + a;
    float* out = P.out;
    for (int d = 0; d < 3; ++d)
      out[g * 3 + d] = tanh_f(2.f * mg * (vx[d] / nrm));
    float ls[3] = {l0, l1, l2};
    for (int j = 0; j < 3; ++j)
      out[(size_t)3 * NBN + g * 3 + j] = fminf(2.f, fmaxf(-5.f, ls[j]));
    out[(size_t)6 * NBN + g] = vl;
  }
}

// ---------------- host entry ----------------
// d_in order (jax pytree / alphabetical flatten):
//  0 obs | 1 conv_b 2 conv_k | 3..8 ego_mlp(b1,b2,b3,w1,w2,w3) | 9 gate_b 10 gate_w
//  11..16 h_mlp | 17 ln1_b 18 ln1_s 19 ln2_b 20 ln2_s | 21..26 log_mlp
//  27 lstd_b 28 lstd_w 29 mag_b 30 mag_w 31 pair_b 32 pair_w
//  33..40 pool_mha(bk,bo,bq,bv,wk,wo,wq,wv) | 41 seed | 42..49 stx_mha
//  50..55 stx_mlp | 56 ter_b 57 ter_w 58 tok_b 59 tok_w 60 val_b 61 val_w
extern "C" void kernel_launch(void* const* d_in, const int* in_sizes, int n_in,
                              void* d_out, int out_size, void* d_ws, size_t ws_size,
                              hipStream_t stream) {
  (void)in_sizes; (void)n_in; (void)out_size; (void)ws_size;
  #define FIN(i) ((const float*)d_in[(i)])
  unsigned short* Wws = (unsigned short*)d_ws;
  float* poolq = (float*)((char*)d_ws + POOLQ_BYTE_OFF);

  PrepArgs pa;
  pa.tok_w = FIN(59); pa.s_wq = FIN(48); pa.s_wk = FIN(46); pa.s_wv = FIN(49); pa.s_wo = FIN(47);
  pa.m1w1 = FIN(53); pa.m1w2 = FIN(54); pa.m1w3 = FIN(55);
  pa.ew1 = FIN(6); pa.ew2 = FIN(7); pa.ew3 = FIN(8);
  pa.lw1 = FIN(24); pa.lw2 = FIN(25);
  pa.p_wk = FIN(37); pa.p_wv = FIN(40); pa.p_wo = FIN(38); pa.p_wq = FIN(39);
  pa.p_bq = FIN(35); pa.seed = FIN(41);
  pa.terw = FIN(57); pa.pairw = FIN(32); pa.hw1 = FIN(14); pa.hw2 = FIN(15); pa.hw3 = FIN(16);
  pa.W = Wws; pa.poolq = poolq;
  hipLaunchKernelGGL(swarm_prep_kernel, dim3(64), dim3(256), 0, stream, pa);

  KArgs ka;
  ka.obs = FIN(0); ka.W = Wws; ka.poolq = poolq;
  ka.tok_b = FIN(58);
  ka.ln1_s = FIN(18); ka.ln1_b = FIN(17); ka.ln2_s = FIN(20); ka.ln2_b = FIN(19);
  ka.sbq = FIN(44); ka.sbk = FIN(42); ka.sbv = FIN(45); ka.sbo = FIN(43);
  ka.m1b1 = FIN(50); ka.m1b2 = FIN(51); ka.m1b3 = FIN(52);
  ka.eb1 = FIN(3); ka.eb2 = FIN(4); ka.eb3 = FIN(5);
  ka.lb1 = FIN(21); ka.lb2 = FIN(22); ka.lb3 = FIN(23); ka.lw3 = FIN(26);
  ka.pbk = FIN(33); ka.pbv = FIN(36); ka.pbo = FIN(34);
  ka.conv_k = FIN(2); ka.conv_b = FIN(1); ka.ter_b = FIN(56);
  ka.pair_b = FIN(31);
  ka.hb1 = FIN(11); ka.hb2 = FIN(12); ka.hb3 = FIN(13);
  ka.gate_w = FIN(10); ka.gate_b = FIN(9);
  ka.mag_w = FIN(30); ka.mag_b = FIN(29);
  ka.lstd_w = FIN(28); ka.lstd_b = FIN(27);
  ka.val_w = FIN(61); ka.val_b = FIN(60);
  ka.out = (float*)d_out;
  hipLaunchKernelGGL(swarm_actor_critic_kernel, dim3(NBN / 16), dim3(128), 0, stream, ka);
  #undef FIN
}
